// ScaledDotProductAttention_764504179346
// MI455X (gfx1250) — compile-verified, matrix operand order checked by reading, not run on hardware
//
#include <hip/hip_runtime.h>
#include <hip/hip_bf16.h>
#include <stdint.h>
#include <math.h>

typedef __attribute__((ext_vector_type(16))) _Float16 v16h;
typedef __attribute__((ext_vector_type(8)))  float    v8f;

#define BATCH 4
#define HEADS 12
#define S_LEN 2048
#define DK    64
#define NW    4                    // waves per workgroup
#define KT_TOT (S_LEN / 16)        // 128 k-tiles
#define KT_PW  (KT_TOT / NW)       // 32 k-tiles per wave
#define KPAD  (DK + 4)             // padded K-tile row (bank skew)
#define SPAD  (32 + 4)             // padded stage row

// ---------------- WMMA helpers ----------------

__device__ __forceinline__ v8f wmma_f16(const v16h& a, const v16h& b, const v8f& c) {
    return __builtin_amdgcn_wmma_f32_16x16x32_f16(false, a, false, b,
                                                  (short)0, c, false, false);
}

// Build the two K=32 B-fragments for a 64-deep contraction from a row-major
// 16x64 tile at `base` with row stride `stride` (works for LDS or global).
__device__ __forceinline__ void build_bfrag(const float* base, int stride,
                                            int n, int h, v16h& b0, v16h& b1) {
    const float* row = base + (size_t)n * stride;     // lane n -> B column n
#pragma unroll
    for (int v = 0; v < 8; ++v) {
        const int k0 = (v & 3) * 2 + h * 8 + (v >> 2) * 16;
        b0[2 * v]     = (_Float16)row[k0];
        b0[2 * v + 1] = (_Float16)row[k0 + 1];
        b1[2 * v]     = (_Float16)row[32 + k0];
        b1[2 * v + 1] = (_Float16)row[32 + k0 + 1];
    }
}

// QK^T tile -> exp(score)*mask in C-fragment layout.
__device__ __forceinline__ v8f exp_mask_tile(const v16h* aq, const v16h& b0, const v16h& b1,
                                             const float* __restrict__ Mb,
                                             int kt, int n, int h, int q0) {
    v8f acc = {};
    acc = wmma_f16(aq[0], b0, acc);
    acc = wmma_f16(aq[1], b1, acc);
    v8f sc;
#pragma unroll
    for (int r = 0; r < 8; ++r) {
        const int qrow = q0 + r + 8 * h;              // C/D: M = r + 8*(lane>>4)
        sc[r] = __expf(acc[r]) * Mb[(size_t)qrow * S_LEN + kt * 16 + n];
    }
    return sc;
}

// Async-DMA one K tile (16 rows x 64 f32, rows contiguous in global) into a
// padded LDS buffer: 2 rows per b128 op, 8 ops, 512B each, fully coalesced.
__device__ __forceinline__ void copy_k_tile_async(float (*dst)[KPAD],
                                                  const float* gsrc, int lane) {
    const int half = lane >> 4;        // which of 2 rows this instruction
    const int g    = lane & 15;        // 16B granule within a 256B row
#pragma unroll
    for (int i = 0; i < 8; ++i) {
        const int row = i * 2 + half;
        const uint32_t lds = (uint32_t)(uintptr_t)&dst[row][g * 4];
        const float* ga = gsrc + row * DK + g * 4;
        asm volatile("global_load_async_to_lds_b128 %0, %1, off"
                     :: "v"(lds), "v"(ga) : "memory");
    }
}

__global__ __launch_bounds__(NW * 32)
void sdpa_wmma_kernel(const float* __restrict__ Q,
                      const float* __restrict__ K,
                      const float* __restrict__ V,
                      const float* __restrict__ mask,
                      float* __restrict__ ctx_out,
                      float* __restrict__ attn_out) {
    const int tid  = threadIdx.x;
    const int wave = tid >> 5;
    const int lane = tid & 31;
    const int qt   = blockIdx.x;          // 0..127  (q tile)
    const int bh   = blockIdx.y;          // 0..47   (b*H + h)
    const int b    = bh / HEADS;
    const int q0   = qt * 16;

    const float* Qb = Q + (size_t)bh * S_LEN * DK;
    const float* Kb = K + (size_t)bh * S_LEN * DK;
    const float* Vb = V + (size_t)bh * S_LEN * DK;
    const float* Mb = mask + (size_t)b * S_LEN * S_LEN;
    float* attn_b   = attn_out + ((size_t)bh * S_LEN + q0) * S_LEN;

    __shared__ float Qs[16][KPAD];
    __shared__ float Ks[NW][2][16][KPAD];     // double-buffered async K tiles
    __shared__ float stage[NW][2][16][SPAD];  // double-buffered normalized attn
    __shared__ float rowsum[16];
    __shared__ float rowinv[16];
    __shared__ float ctxred[16][DK];

    // ---- load Q tile (contiguous, coalesced) + init reductions ----
    for (int i = tid; i < 16 * DK; i += NW * 32)
        Qs[i / DK][i % DK] = Qb[(size_t)q0 * DK + i];
    for (int i = tid; i < 16 * DK; i += NW * 32)
        ctxred[i / DK][i % DK] = 0.0f;
    if (tid < 16) rowsum[tid] = 0.0f;
    __syncthreads();

    // ---- build Q A-fragments (1/sqrt(64) folded in) ----
    const int n = lane & 15;
    const int h = lane >> 4;
    const float scale = 0.125f;
    v16h aq[2];
#pragma unroll
    for (int v = 0; v < 8; ++v) {
        const int k0 = (v & 3) * 2 + h * 8 + (v >> 2) * 16;
        aq[0][2 * v]     = (_Float16)(Qs[n][k0]          * scale);
        aq[0][2 * v + 1] = (_Float16)(Qs[n][k0 + 1]      * scale);
        aq[1][2 * v]     = (_Float16)(Qs[n][32 + k0]     * scale);
        aq[1][2 * v + 1] = (_Float16)(Qs[n][32 + k0 + 1] * scale);
    }

    // ---- pass 1: row sums of exp(scores)*mask over this wave's k range ----
    // K tiles are DMA'd into double-buffered LDS; next tile's async copy
    // overlaps this tile's WMMA/exp (s_wait_asynccnt 8 = previous buffer done).
    const int kt_begin = wave * KT_PW;
    float rsum[8] = {0.f, 0.f, 0.f, 0.f, 0.f, 0.f, 0.f, 0.f};

    copy_k_tile_async(Ks[wave][0], Kb + (size_t)kt_begin * 16 * DK, lane);
    for (int it = 0; it < KT_PW; ++it) {
        const int kt = kt_begin + it;
        const int p  = it & 1;
        if (it + 1 < KT_PW) {
            copy_k_tile_async(Ks[wave][p ^ 1], Kb + (size_t)(kt + 1) * 16 * DK, lane);
            asm volatile("s_wait_asynccnt 0x8" ::: "memory");   // buffer p landed
        } else {
            asm volatile("s_wait_asynccnt 0x0" ::: "memory");
        }
        __builtin_prefetch(Mb + (size_t)(q0 + lane) * S_LEN + (kt + 1) * 16, 0, 0);
        v16h bk0, bk1;
        build_bfrag(&Ks[wave][p][0][0], KPAD, n, h, bk0, bk1);  // ds_load path
        v8f sc = exp_mask_tile(aq, bk0, bk1, Mb, kt, n, h, q0);
#pragma unroll
        for (int r = 0; r < 8; ++r) rsum[r] += sc[r];
    }
#pragma unroll
    for (int r = 0; r < 8; ++r)
        atomicAdd(&rowsum[r + 8 * h], rsum[r]);                 // ds_add_f32
    __syncthreads();
    if (tid < 16) rowinv[tid] = 1.0f / (rowsum[tid] + 1e-8f);
    __syncthreads();

    // ---- pass 2: recompute, normalize, async-store attn, and attn @ V ----
    // In this phase ASYNCcnt carries only attn stores (stores complete in
    // order among stores), so s_wait_asynccnt 4 frees stage buffer j-2.
    v8f ctx[4] = {};                               // 4 N-tiles of 16 cols
    for (int j = 0; j < KT_PW / 2; ++j) {
        const int ktA = kt_begin + 2 * j;
        const int p2  = j & 1;
        if (j >= 2)
            asm volatile("s_wait_asynccnt 0x4" ::: "memory");

#pragma unroll
        for (int t = 0; t < 2; ++t) {
            const int kt = ktA + t;
            v16h bk0, bk1;
            build_bfrag(Kb + (size_t)kt * 16 * DK, DK, n, h, bk0, bk1);  // L2-resident
            v8f sc = exp_mask_tile(aq, bk0, bk1, Mb, kt, n, h, q0);
#pragma unroll
            for (int r = 0; r < 8; ++r) {
                const int Mc = r + 8 * h;
                stage[wave][p2][Mc][t * 16 + n] = sc[r] * rowinv[Mc];
            }
        }
        // staged LDS stores must land before the async engine reads them
        asm volatile("s_wait_dscnt 0" ::: "memory");

        // attn tile out: 16 rows x 32 contiguous floats -> 4 b128 async stores
        {
            const int row4 = lane >> 3;            // 4 rows per instruction
            const int g    = lane & 7;             // 16B granule within 128B row
#pragma unroll
            for (int i = 0; i < 4; ++i) {
                const int row = i * 4 + row4;
                const uint32_t lds = (uint32_t)(uintptr_t)&stage[wave][p2][row][g * 4];
                float* ga = attn_b + (size_t)row * S_LEN + ktA * 16 + g * 4;
                asm volatile("global_store_async_from_lds_b128 %0, %1, off"
                             :: "v"(ga), "v"(lds) : "memory");
            }
        }

        // rebuild staged 16x32 attn tile as an A-fragment (LDS reads only)
        v16h aa;
#pragma unroll
        for (int v = 0; v < 8; ++v) {
            const int k0 = (v & 3) * 2 + h * 8 + (v >> 2) * 16;
            aa[2 * v]     = (_Float16)stage[wave][p2][n][k0];
            aa[2 * v + 1] = (_Float16)stage[wave][p2][n][k0 + 1];
        }
        // V B-fragments: rows kbase..kbase+31 (L2-resident), one N-tile per wmma
        const int kbase = ktA * 16;
#pragma unroll
        for (int nt = 0; nt < 4; ++nt) {
            v16h bv;
#pragma unroll
            for (int v = 0; v < 8; ++v) {
                const int k0 = (v & 3) * 2 + h * 8 + (v >> 2) * 16;
                const float* Vr = Vb + (size_t)(kbase + k0) * DK + nt * 16 + n;
                bv[2 * v]     = (_Float16)Vr[0];
                bv[2 * v + 1] = (_Float16)Vr[DK];
            }
            ctx[nt] = wmma_f16(aa, bv, ctx[nt]);
        }
    }

    // ---- cross-wave context reduction (each wave covered 512 of 2048 k) ----
#pragma unroll
    for (int nt = 0; nt < 4; ++nt)
#pragma unroll
        for (int r = 0; r < 8; ++r)
            atomicAdd(&ctxred[r + 8 * h][nt * 16 + n], ctx[nt][r]);   // ds_add_f32
    __syncthreads();

    float* ctx_b = ctx_out + ((size_t)bh * S_LEN + q0) * DK;
    for (int i = tid; i < 16 * DK; i += NW * 32)
        ctx_b[i] = ctxred[i / DK][i % DK];
    // S_ENDPGM performs an implicit wait-idle, draining outstanding async stores.
}

extern "C" void kernel_launch(void* const* d_in, const int* in_sizes, int n_in,
                              void* d_out, int out_size, void* d_ws, size_t ws_size,
                              hipStream_t stream) {
    const float* Q    = (const float*)d_in[0];
    const float* K    = (const float*)d_in[1];
    const float* V    = (const float*)d_in[2];
    const float* mask = (const float*)d_in[3];

    float* ctx_out  = (float*)d_out;                                   // (B,H,S,DK)
    float* attn_out = ctx_out + (size_t)BATCH * HEADS * S_LEN * DK;    // (B,H,S,S)

    dim3 grid(S_LEN / 16, BATCH * HEADS);   // 128 q-tiles x 48 (b,h)
    dim3 block(NW * 32);                    // 4 waves
    sdpa_wmma_kernel<<<grid, block, 0, stream>>>(Q, K, V, mask, ctx_out, attn_out);
}
